// GCN_84207128805414
// MI455X (gfx1250) — matrix operand order repaired and checked by reading, not compiled, not run on hardware
//
#include <hip/hip_runtime.h>

typedef __attribute__((ext_vector_type(16))) _Float16 v16h;
typedef __attribute__((ext_vector_type(8)))  float    v8f;

#define N_NODES 50000
#define N_EDGES 800000
#define IN_F    128
#define HID     256   // HEADS*OUT_FEAT
#define HEADS   4
#define DFEAT   64
#define NCLS    40
#define NCLSP   48    // padded to multiple of 16

// ---------- helpers ----------
__device__ __forceinline__ unsigned fenc(float x) {
    unsigned u = __float_as_uint(x);
    return (u & 0x80000000u) ? ~u : (u | 0x80000000u);
}
__device__ __forceinline__ float fdec(unsigned k) {
    unsigned u = (k & 0x80000000u) ? (k & 0x7fffffffu) : ~k;
    return __uint_as_float(u);
}

struct alignas(16) H8 { _Float16 h[8]; };

// ---------- conversion ----------
__global__ void f32_to_f16_k(const float* __restrict__ in, _Float16* __restrict__ out, int n) {
    int i = blockIdx.x * blockDim.x + threadIdx.x;
    if (i < n) out[i] = (_Float16)in[i];
}

// ---------- pre-swizzle W (row-major f32 [K][Ncols]) into WMMA B-fragment layout ----------
// frag[(kt*coltiles + ct)*512 + lane*16 + j] = B[kt*32 + koff(lane,j)][ct*16 + (lane&15)]
// with koff = (j>=8 ? 16 : 0) + (lane>=16 ? 8 : 0) + (j&7).  Zero-pads columns >= Ncols.
__global__ void build_bfrag_k(const float* __restrict__ W, _Float16* __restrict__ frag,
                              int Ncols, int coltiles, int total) {
    int i = blockIdx.x * blockDim.x + threadIdx.x;
    if (i >= total) return;
    int j    = i & 15;
    int lane = (i >> 4) & 31;
    int f    = i >> 9;                       // fragment index
    int kt   = f / coltiles;
    int ct   = f - kt * coltiles;
    int k    = kt * 32 + ((j >> 3) << 4) + ((lane >> 4) << 3) + (j & 7);
    int n    = ct * 16 + (lane & 15);
    frag[i] = (n < Ncols) ? (_Float16)W[(size_t)k * Ncols + n] : (_Float16)0.f;
}

// ---------- WMMA GEMM: C[M][ldc] = A[M][KDIM] * B[KDIM][16*totct] ----------
// One wave computes a 16 x (16*CT) strip: A fragment loaded once per k-step,
// B fragments come pre-swizzled (two b128 loads each).
template <int KDIM, int CT>
__global__ void gemm_wmma_f16(const _Float16* __restrict__ A,
                              const _Float16* __restrict__ Bfrag,
                              float* __restrict__ C,
                              int ldc, int totct, int colgroups, int ntiles) {
    const int wave = threadIdx.x >> 5;
    const int lane = threadIdx.x & 31;
    const int tile = blockIdx.x * (blockDim.x >> 5) + wave;
    if (tile >= ntiles) return;                 // wave-uniform: EXEC stays all-ones
    const int rt = tile / colgroups;
    const int cg = tile - rt * colgroups;
    const int row0 = rt << 4;
    const int l15  = lane & 15;
    const int hi   = lane >> 4;                 // 0 or 1 (lane half)
    const int aoff = hi * 8;                    // A: lanes 16-31 hold K+8

    const _Float16* arow = A + (size_t)(row0 + l15) * (size_t)KDIM;
    v8f acc[CT];
#pragma unroll
    for (int c = 0; c < CT; ++c) acc[c] = (v8f){};

#pragma unroll
    for (int k0 = 0; k0 < KDIM; k0 += 32) {
        // A fragment: two contiguous 8-half (16B) runs per lane
        H8 a0 = *reinterpret_cast<const H8*>(arow + k0 + aoff);
        H8 a1 = *reinterpret_cast<const H8*>(arow + k0 + 16 + aoff);
        v16h av;
#pragma unroll
        for (int j = 0; j < 8; ++j) { av[j] = a0.h[j]; av[8 + j] = a1.h[j]; }

#pragma unroll
        for (int c = 0; c < CT; ++c) {
            // pre-swizzled B fragment: 16 contiguous halves per lane (2 x b128)
            const _Float16* bp = Bfrag +
                ((((size_t)(k0 >> 5) * totct + (cg * CT + c)) << 5) + lane) * 16;
            H8 b0 = *reinterpret_cast<const H8*>(bp);
            H8 b1 = *reinterpret_cast<const H8*>(bp + 8);
            v16h bv;
#pragma unroll
            for (int j = 0; j < 8; ++j) { bv[j] = b0.h[j]; bv[8 + j] = b1.h[j]; }

            acc[c] = __builtin_amdgcn_wmma_f32_16x16x32_f16(
                false, av, false, bv, (short)0, acc[c], false, false);
        }
    }

    const int mbase = hi * 8;
#pragma unroll
    for (int c = 0; c < CT; ++c) {
        const int col0 = (cg * CT + c) << 4;
#pragma unroll
        for (int r = 0; r < 8; ++r)
            C[(size_t)(row0 + mbase + r) * (size_t)ldc + col0 + l15] = acc[c][r];
    }
}

// ---------- attention coefficients: el/er = sum_d h*al / h*ar ----------
__global__ void attn_coef_k(const float* __restrict__ h,
                            const float* __restrict__ al, const float* __restrict__ ar,
                            float* __restrict__ el, float* __restrict__ er,
                            int total, int D, int heads, int ldh) {
    int i = blockIdx.x * blockDim.x + threadIdx.x;
    if (i >= total) return;
    int n = i / heads, hh = i - n * heads;
    const float* hp = h + (size_t)n * ldh + hh * D;
    const float* alp = al + hh * D;
    const float* arp = ar + hh * D;
    float sl = 0.f, sr = 0.f;
    for (int d = 0; d < D; ++d) { float v = hp[d]; sl += v * alp[d]; sr += v * arp[d]; }
    el[i] = sl; er[i] = sr;
}

// ---------- layer-1 edge passes (H=4, vectorized float4) ----------
__global__ void edge1_max_k(const int* __restrict__ src, const int* __restrict__ dst,
                            const float* __restrict__ el, const float* __restrict__ er,
                            float* __restrict__ e_out, unsigned* __restrict__ emax, int E) {
    int i = blockIdx.x * blockDim.x + threadIdx.x;
    if (i >= E) return;
    int s = src[i], d = dst[i];
    float4 L = *reinterpret_cast<const float4*>(el + 4 * (size_t)s);
    float4 R = *reinterpret_cast<const float4*>(er + 4 * (size_t)d);
    float v[4] = {L.x + R.x, L.y + R.y, L.z + R.z, L.w + R.w};
#pragma unroll
    for (int hh = 0; hh < 4; ++hh) {
        float e = v[hh] > 0.f ? v[hh] : 0.2f * v[hh];    // LeakyReLU(0.2)
        e_out[(size_t)i * 4 + hh] = e;
        atomicMax(&emax[(size_t)d * 4 + hh], fenc(e));
    }
}

__global__ void edge1_sum_k(const int* __restrict__ dst,
                            const unsigned* __restrict__ emax,
                            float* __restrict__ e_io, float* __restrict__ denom, int E) {
    int i = blockIdx.x * blockDim.x + threadIdx.x;
    if (i >= E) return;
    int d = dst[i];
#pragma unroll
    for (int hh = 0; hh < 4; ++hh) {
        unsigned k = emax[(size_t)d * 4 + hh];
        float m = (k == 0u) ? 0.f : fdec(k);             // isolated -> 0 (ref semantics)
        float ee = expf(e_io[(size_t)i * 4 + hh] - m);
        e_io[(size_t)i * 4 + hh] = ee;
        atomicAdd(&denom[(size_t)d * 4 + hh], ee);
    }
}

// finalize alpha = ee / max(denom,1) per edge (removes denom gather from hot agg kernel)
__global__ void alpha1_k(const int* __restrict__ dst, const float* __restrict__ denom,
                         float* __restrict__ e_io, int E) {
    int i = blockIdx.x * blockDim.x + threadIdx.x;
    if (i >= E) return;
    int d = dst[i];
    float4 dn = *reinterpret_cast<const float4*>(denom + 4 * (size_t)d);
    float4 ee = *reinterpret_cast<const float4*>(e_io + 4 * (size_t)i);
    float4 a;
    a.x = ee.x / (dn.x > 0.f ? dn.x : 1.f);
    a.y = ee.y / (dn.y > 0.f ? dn.y : 1.f);
    a.z = ee.z / (dn.z > 0.f ? dn.z : 1.f);
    a.w = ee.w / (dn.w > 0.f ? dn.w : 1.f);
    *reinterpret_cast<float4*>(e_io + 4 * (size_t)i) = a;
}

// 16 threads per edge: thread q covers float4 chunk q of all 4 heads
__global__ void edge1_agg_k(const int* __restrict__ src, const int* __restrict__ dst,
                            const float* __restrict__ alpha,
                            const float* __restrict__ h1, float* __restrict__ out1, int total) {
    int t = blockIdx.x * blockDim.x + threadIdx.x;
    if (t >= total) return;
    int e = t >> 4;
    int q = t & 15;                // float4 chunk within D=64
    int s = src[e], d = dst[e];
    float4 a4 = *reinterpret_cast<const float4*>(alpha + 4 * (size_t)e);
    float a[4] = {a4.x, a4.y, a4.z, a4.w};
    const float* hp = h1 + (size_t)s * HID + q * 4;
    float* op = out1 + (size_t)d * HID + q * 4;
#pragma unroll
    for (int hh = 0; hh < 4; ++hh) {
        float4 hv = *reinterpret_cast<const float4*>(hp + hh * DFEAT);
        atomicAdd(op + hh * DFEAT + 0, hv.x * a[hh]);
        atomicAdd(op + hh * DFEAT + 1, hv.y * a[hh]);
        atomicAdd(op + hh * DFEAT + 2, hv.z * a[hh]);
        atomicAdd(op + hh * DFEAT + 3, hv.w * a[hh]);
    }
}

// ---------- bias + ELU + convert to f16 for GEMM2 ----------
__global__ void bias_elu_k(const float* __restrict__ acc, const float* __restrict__ b1,
                           _Float16* __restrict__ hin, int total) {
    int i = blockIdx.x * blockDim.x + threadIdx.x;
    if (i >= total) return;
    int c = i & (HID - 1);
    float v = acc[i] + b1[c];
    v = v > 0.f ? v : (expf(v) - 1.f);                   // ELU
    hin[i] = (_Float16)v;
}

// ---------- layer-2 edge passes (H=1) ----------
__global__ void edge2_max_k(const int* __restrict__ src, const int* __restrict__ dst,
                            const float* __restrict__ el, const float* __restrict__ er,
                            float* __restrict__ e_out, unsigned* __restrict__ emax, int E) {
    int i = blockIdx.x * blockDim.x + threadIdx.x;
    if (i >= E) return;
    int s = src[i], d = dst[i];
    float v = el[s] + er[d];
    float e = v > 0.f ? v : 0.2f * v;
    e_out[i] = e;
    atomicMax(&emax[d], fenc(e));
}

__global__ void edge2_sum_k(const int* __restrict__ dst, const unsigned* __restrict__ emax,
                            float* __restrict__ e_io, float* __restrict__ denom, int E) {
    int i = blockIdx.x * blockDim.x + threadIdx.x;
    if (i >= E) return;
    int d = dst[i];
    unsigned k = emax[d];
    float m = (k == 0u) ? 0.f : fdec(k);
    float ee = expf(e_io[i] - m);
    e_io[i] = ee;
    atomicAdd(&denom[d], ee);
}

__global__ void alpha2_k(const int* __restrict__ dst, const float* __restrict__ denom,
                         float* __restrict__ e_io, int E) {
    int i = blockIdx.x * blockDim.x + threadIdx.x;
    if (i >= E) return;
    float dn = denom[dst[i]];
    e_io[i] = e_io[i] / (dn > 0.f ? dn : 1.f);
}

__global__ void edge2_agg_k(const int* __restrict__ src, const int* __restrict__ dst,
                            const float* __restrict__ alpha,
                            const float* __restrict__ h2, float* __restrict__ out2, int total) {
    int t = blockIdx.x * blockDim.x + threadIdx.x;
    if (t >= total) return;
    int e = t / 10;
    int q = t - e * 10;            // float4 chunk within C=40
    int s = src[e], d = dst[e];
    float a = alpha[e];
    const float4 hv = *reinterpret_cast<const float4*>(h2 + (size_t)s * NCLSP + q * 4);
    float* op = out2 + (size_t)d * NCLS + q * 4;
    atomicAdd(op + 0, hv.x * a);
    atomicAdd(op + 1, hv.y * a);
    atomicAdd(op + 2, hv.z * a);
    atomicAdd(op + 3, hv.w * a);
}

__global__ void final_bias_k(const float* __restrict__ acc, const float* __restrict__ b2,
                             float* __restrict__ out, int total) {
    int i = blockIdx.x * blockDim.x + threadIdx.x;
    if (i >= total) return;
    int c = i % NCLS;
    out[i] = acc[i] + b2[c];
}

// ---------- host ----------
extern "C" void kernel_launch(void* const* d_in, const int* in_sizes, int n_in,
                              void* d_out, int out_size, void* d_ws, size_t ws_size,
                              hipStream_t stream) {
    const float* x   = (const float*)d_in[0];
    const int*   src = (const int*)d_in[1];
    const int*   dst = (const int*)d_in[2];
    const float* W1  = (const float*)d_in[3];
    const float* al1 = (const float*)d_in[4];
    const float* ar1 = (const float*)d_in[5];
    const float* b1  = (const float*)d_in[6];
    const float* W2  = (const float*)d_in[7];
    const float* al2 = (const float*)d_in[8];
    const float* ar2 = (const float*)d_in[9];
    const float* b2  = (const float*)d_in[10];
    float* out = (float*)d_out;
    (void)in_sizes; (void)n_in; (void)out_size; (void)ws_size;

    char* ws = (char*)d_ws;
    size_t off = 0;
    auto alloc = [&](size_t bytes) -> char* {
        char* p = ws + off;
        off += (bytes + 255) & ~(size_t)255;
        return p;
    };

    _Float16* xf16  = (_Float16*)alloc((size_t)N_NODES * IN_F * 2);
    _Float16* w1frag= (_Float16*)alloc((size_t)(IN_F / 32) * (HID / 16) * 512 * 2);
    _Float16* w2frag= (_Float16*)alloc((size_t)(HID / 32) * (NCLSP / 16) * 512 * 2);
    float*    h1    = (float*)alloc((size_t)N_NODES * HID * 4);
    float*    el1   = (float*)alloc((size_t)N_NODES * HEADS * 4);
    float*    er1   = (float*)alloc((size_t)N_NODES * HEADS * 4);
    unsigned* emax1 = (unsigned*)alloc((size_t)N_NODES * HEADS * 4);
    float*    den1  = (float*)alloc((size_t)N_NODES * HEADS * 4);
    float*    e1    = (float*)alloc((size_t)N_EDGES * HEADS * 4);
    float*    out1  = (float*)alloc((size_t)N_NODES * HID * 4);
    _Float16* hin   = (_Float16*)alloc((size_t)N_NODES * HID * 2);
    float*    h2    = (float*)alloc((size_t)N_NODES * NCLSP * 4);
    float*    el2   = (float*)alloc((size_t)N_NODES * 4);
    float*    er2   = (float*)alloc((size_t)N_NODES * 4);
    unsigned* emax2 = (unsigned*)alloc((size_t)N_NODES * 4);
    float*    den2  = (float*)alloc((size_t)N_NODES * 4);
    float*    e2    = (float*)alloc((size_t)N_EDGES * 4);
    float*    out2  = (float*)alloc((size_t)N_NODES * NCLS * 4);

    const int B = 256;

    // zero accumulators / max-encodings (enc==0 means "no edge")
    hipMemsetAsync(emax1, 0, (size_t)N_NODES * HEADS * 4, stream);
    hipMemsetAsync(den1,  0, (size_t)N_NODES * HEADS * 4, stream);
    hipMemsetAsync(out1,  0, (size_t)N_NODES * HID * 4, stream);
    hipMemsetAsync(emax2, 0, (size_t)N_NODES * 4, stream);
    hipMemsetAsync(den2,  0, (size_t)N_NODES * 4, stream);
    hipMemsetAsync(out2,  0, (size_t)N_NODES * NCLS * 4, stream);

    // operand prep: x -> f16, W1/W2 -> pre-swizzled B-fragments (with f32->f16 + padding)
    f32_to_f16_k<<<((size_t)N_NODES * IN_F + B - 1) / B, B, 0, stream>>>(x, xf16, N_NODES * IN_F);
    {
        int tot1 = (IN_F / 32) * (HID / 16) * 512;
        build_bfrag_k<<<(tot1 + B - 1) / B, B, 0, stream>>>(W1, w1frag, HID, HID / 16, tot1);
        int tot2 = (HID / 32) * (NCLSP / 16) * 512;
        build_bfrag_k<<<(tot2 + B - 1) / B, B, 0, stream>>>(W2, w2frag, NCLS, NCLSP / 16, tot2);
    }

    // GEMM1: 3125 row-tiles x 4 col-groups (CT=4 -> 16x64 strip per wave)
    {
        int ntiles = (N_NODES / 16) * (HID / 64);
        gemm_wmma_f16<IN_F, 4><<<(ntiles + 7) / 8, B, 0, stream>>>(
            xf16, w1frag, h1, HID, HID / 16, HID / 64, ntiles);
    }

    attn_coef_k<<<(N_NODES * HEADS + B - 1) / B, B, 0, stream>>>(
        h1, al1, ar1, el1, er1, N_NODES * HEADS, DFEAT, HEADS, HID);

    edge1_max_k<<<(N_EDGES + B - 1) / B, B, 0, stream>>>(src, dst, el1, er1, e1, emax1, N_EDGES);
    edge1_sum_k<<<(N_EDGES + B - 1) / B, B, 0, stream>>>(dst, emax1, e1, den1, N_EDGES);
    alpha1_k<<<(N_EDGES + B - 1) / B, B, 0, stream>>>(dst, den1, e1, N_EDGES);
    edge1_agg_k<<<((size_t)N_EDGES * 16 + B - 1) / B, B, 0, stream>>>(
        src, dst, e1, h1, out1, N_EDGES * 16);

    bias_elu_k<<<((size_t)N_NODES * HID + B - 1) / B, B, 0, stream>>>(out1, b1, hin, N_NODES * HID);

    // GEMM2: 3125 row-tiles, CT=3 (one 16x48 strip per wave, cols 40..47 are pad)
    {
        int ntiles = (N_NODES / 16);
        gemm_wmma_f16<HID, 3><<<(ntiles + 7) / 8, B, 0, stream>>>(
            hin, w2frag, h2, NCLSP, NCLSP / 16, 1, ntiles);
    }

    attn_coef_k<<<(N_NODES + B - 1) / B, B, 0, stream>>>(
        h2, al2, ar2, el2, er2, N_NODES, NCLS, 1, NCLSP);

    edge2_max_k<<<(N_EDGES + B - 1) / B, B, 0, stream>>>(src, dst, el2, er2, e2, emax2, N_EDGES);
    edge2_sum_k<<<(N_EDGES + B - 1) / B, B, 0, stream>>>(dst, emax2, e2, den2, N_EDGES);
    alpha2_k<<<(N_EDGES + B - 1) / B, B, 0, stream>>>(dst, den2, e2, N_EDGES);
    edge2_agg_k<<<((size_t)N_EDGES * 10 + B - 1) / B, B, 0, stream>>>(
        src, dst, e2, h2, out2, N_EDGES * 10);

    final_bias_k<<<((size_t)N_NODES * NCLS + B - 1) / B, B, 0, stream>>>(out2, b2, out, N_NODES * NCLS);
}
